// StandardAttention_8280696947021
// MI455X (gfx1250) — matrix operand order repaired and checked by reading, not compiled, not run on hardware
//
#include <hip/hip_runtime.h>

#define B_  2
#define T_  2048
#define H_  1024
#define NH_ 16
#define HD_ 64
#define K_  H_     // GEMM reduction dim for all projections

typedef __bf16 bf16;
typedef __attribute__((ext_vector_type(16))) __bf16 v16bf;
typedef __attribute__((ext_vector_type(8)))  __bf16 v8bf;
typedef __attribute__((ext_vector_type(8)))  float  v8f;

union BF16x16 { v16bf v; v8bf h[2]; };

// wave-local LDS ordering (per-wave P tile; DS ops are in-order within a wave)
#if defined(__has_builtin)
#if __has_builtin(__builtin_amdgcn_s_wait_dscnt)
#define WAVE_LDS_SYNC() do { asm volatile("" ::: "memory"); \
                             __builtin_amdgcn_s_wait_dscnt(0); \
                             asm volatile("" ::: "memory"); } while (0)
#endif
#endif
#ifndef WAVE_LDS_SYNC
#define WAVE_LDS_SYNC() __syncthreads()
#endif

__device__ __forceinline__ bf16 f2bf(float f) {
  union { float f; unsigned u; } c; c.f = f;
  unsigned r = c.u + 0x7fffu + ((c.u >> 16) & 1u);   // round-to-nearest-even
  unsigned short s = (unsigned short)(r >> 16);
  union { unsigned short s; bf16 b; } o; o.s = s; return o.b;
}

__device__ __forceinline__ v8f wmma_bf16(const BF16x16& a, const BF16x16& b, v8f c) {
  return __builtin_amdgcn_wmma_f32_16x16x32_bf16(
      /*neg_a=*/false, a.v, /*neg_b=*/false, b.v,
      /*c_mod=*/(short)0, c, /*reuse_a=*/false, /*reuse_b=*/false);
}

// ---------------- fp32 -> bf16 convert (x) ----------------
__global__ void __launch_bounds__(256) cvt_f32_bf16(const float* __restrict__ in,
                                                    bf16* __restrict__ out, long n) {
  long i = (long)blockIdx.x * blockDim.x + threadIdx.x;
  long stride = (long)gridDim.x * blockDim.x;
  for (; i < n; i += stride) out[i] = f2bf(in[i]);
}

// ---------------- W[H][H] (in,out) -> WT[H][H] bf16 (out,in) ----------------
__global__ void __launch_bounds__(256) transpose_cvt(const float* __restrict__ W,
                                                     bf16* __restrict__ WT) {
  int idx = blockIdx.x * 256 + threadIdx.x;    // over H_*H_
  int o = idx >> 10;          // out-feature (row of WT)
  int i = idx & (H_ - 1);     // in-feature (contiguous in WT)
  WT[idx] = f2bf(W[(size_t)i * H_ + o]);
}

// ============ WMMA GEMM: Y = A(bf16 MxK) @ Wt(bf16 NxK)^T + bias ============
// One wave computes a 32x64 tile (2 M-subtiles x 4 N-subtiles); double-buffered
// k-loop so loads for k+32 are in flight during wmma on k.
struct KChunk { BF16x16 a0, a1, b[4]; };

__device__ __forceinline__ void load_chunk(KChunk& c, const bf16* Arow0,
                                           const bf16* Arow1, const bf16* Bbase,
                                           int kk, int half) {
  c.a0.h[0] = *(const v8bf*)(Arow0 + kk + half * 8);
  c.a0.h[1] = *(const v8bf*)(Arow0 + kk + 16 + half * 8);
  c.a1.h[0] = *(const v8bf*)(Arow1 + kk + half * 8);
  c.a1.h[1] = *(const v8bf*)(Arow1 + kk + 16 + half * 8);
#pragma unroll
  for (int cc = 0; cc < 4; ++cc) {
    const bf16* p = Bbase + (size_t)cc * 16 * K_ + kk;
    c.b[cc].h[0] = *(const v8bf*)(p + half * 8);
    c.b[cc].h[1] = *(const v8bf*)(p + 16 + half * 8);
  }
}

__device__ __forceinline__ void compute_chunk(v8f (&acc)[2][4], const KChunk& c) {
#pragma unroll
  for (int cc = 0; cc < 4; ++cc) {
    acc[0][cc] = wmma_bf16(c.a0, c.b[cc], acc[0][cc]);
    acc[1][cc] = wmma_bf16(c.a1, c.b[cc], acc[1][cc]);
  }
}

// mode: 0 -> bf16 store [B,NH,T,HD]; 1 -> bf16 store [B,NH,HD,T] (V^T);
//       2 -> fp32 store [M,N] (final out)
__global__ void __launch_bounds__(256) proj_gemm(const bf16* __restrict__ A,
                                                 const bf16* __restrict__ Bt,
                                                 const float* __restrict__ bias,
                                                 void* __restrict__ outp,
                                                 int mode) {
  const int wave = threadIdx.x >> 5;
  const int lane = threadIdx.x & 31;
  const int half = lane >> 4;
  const int ln   = lane & 15;
  const int tile = blockIdx.x * 8 + wave;   // (M/32)*(N/64) = 128*16 = 2048 tiles
  const int mt = tile >> 4;
  const int nt = tile & 15;

  const bf16* Arow0 = A + (size_t)(mt * 32 + ln) * K_;
  const bf16* Arow1 = Arow0 + (size_t)16 * K_;
  const bf16* Bbase = Bt + (size_t)(nt * 64 + ln) * K_;

  v8f acc[2][4] = {{{}, {}, {}, {}}, {{}, {}, {}, {}}};

  KChunk c0, c1;
  load_chunk(c0, Arow0, Arow1, Bbase, 0, half);
  for (int kk = 0; kk < K_ - 64; kk += 64) {
    load_chunk(c1, Arow0, Arow1, Bbase, kk + 32, half);
    compute_chunk(acc, c0);
    load_chunk(c0, Arow0, Arow1, Bbase, kk + 64, half);
    compute_chunk(acc, c1);
  }
  load_chunk(c1, Arow0, Arow1, Bbase, K_ - 32, half);
  compute_chunk(acc, c0);
  compute_chunk(acc, c1);

#pragma unroll
  for (int s = 0; s < 2; ++s) {
#pragma unroll
    for (int c = 0; c < 4; ++c) {
      const int n = nt * 64 + c * 16 + ln;
      const float bv = bias[n];
#pragma unroll
      for (int r = 0; r < 8; ++r) {
        const int m = mt * 32 + s * 16 + half * 8 + r;   // C layout: M = r + 8*half
        float val = acc[s][c][r] + bv;
        if (mode == 2) {
          ((float*)outp)[(size_t)m * H_ + n] = val;
        } else {
          const int bb = m / T_, t = m % T_;
          const int hh = n >> 6, d = n & 63;
          size_t idx = (mode == 0)
              ? ((((size_t)bb * NH_ + hh) * T_ + t) * HD_ + d)    // [B,NH,T,HD]
              : ((((size_t)bb * NH_ + hh) * HD_ + d) * T_ + t);   // [B,NH,HD,T]
          ((bf16*)outp)[idx] = f2bf(val);
        }
      }
    }
  }
}

// ============ Flash attention: one wave per (b, h, 16-query tile) ============
struct KTile { BF16x16 b[4]; };   // [0]=tile0/ch0 [1]=tile1/ch0 [2]=tile0/ch1 [3]=tile1/ch1

__device__ __forceinline__ void load_ktile(KTile& kt, const bf16* kbase,
                                           int kb, int ln, int half) {
  const bf16* kr0 = kbase + (size_t)(kb + ln) * HD_ + half * 16;
  const bf16* kr1 = kbase + (size_t)(kb + 16 + ln) * HD_ + half * 16;
  kt.b[0].h[0] = *(const v8bf*)(kr0);      kt.b[0].h[1] = *(const v8bf*)(kr0 + 8);
  kt.b[1].h[0] = *(const v8bf*)(kr1);      kt.b[1].h[1] = *(const v8bf*)(kr1 + 8);
  kt.b[2].h[0] = *(const v8bf*)(kr0 + 32); kt.b[2].h[1] = *(const v8bf*)(kr0 + 40);
  kt.b[3].h[0] = *(const v8bf*)(kr1 + 32); kt.b[3].h[1] = *(const v8bf*)(kr1 + 40);
}

__global__ void __launch_bounds__(256) flash_attn(const bf16* __restrict__ q,
                                                  const bf16* __restrict__ k,
                                                  const bf16* __restrict__ vT,
                                                  const int*  __restrict__ mask,
                                                  bf16* __restrict__ ctx) {
  __shared__ __align__(16) bf16 Plds[8][16 * 32];   // per-wave 16x32 P tile
  const int wave = threadIdx.x >> 5;
  const int lane = threadIdx.x & 31;
  const int half = lane >> 4;
  const int ln   = lane & 15;
  const int tile = blockIdx.x * 8 + wave;           // 4096 tiles
  const int qt = tile & (T_ / 16 - 1);
  const int h  = (tile >> 7) & (NH_ - 1);
  const int b  = tile >> 11;
  const float scale = 0.125f;                       // HD^-0.5

  const bf16* qrow = q + (((size_t)b * NH_ + h) * T_ + qt * 16 + ln) * HD_;
  BF16x16 aq0, aq1;
  aq0.h[0] = *(const v8bf*)(qrow + half * 16);
  aq0.h[1] = *(const v8bf*)(qrow + half * 16 + 8);
  aq1.h[0] = *(const v8bf*)(qrow + 32 + half * 16);
  aq1.h[1] = *(const v8bf*)(qrow + 40 + half * 16);

  const bf16* kbase = k  + ((size_t)b * NH_ + h) * (size_t)T_ * HD_;
  const bf16* vbase = vT + ((size_t)b * NH_ + h) * (size_t)HD_ * T_;
  const int* mrow = mask + b * T_;
  bf16* myP = &Plds[wave][0];

  float m_run[8], l_run[8];
#pragma unroll
  for (int r = 0; r < 8; ++r) { m_run[r] = -1e30f; l_run[r] = 0.f; }
  v8f o[4] = {{}, {}, {}, {}};

  auto body = [&](const KTile& kt, int kb) {
    // ---- scores for 32 keys ----
    v8f s0 = {}, s1 = {};
    s0 = wmma_bf16(aq0, kt.b[0], s0);
    s1 = wmma_bf16(aq0, kt.b[1], s1);
    s0 = wmma_bf16(aq1, kt.b[2], s0);
    s1 = wmma_bf16(aq1, kt.b[3], s1);

    // ---- online softmax (rows live across the 16 lanes of each half) ----
    const float mb0 = (mrow[kb + ln] == 0)      ? -1e30f : 0.f;
    const float mb1 = (mrow[kb + 16 + ln] == 0) ? -1e30f : 0.f;
    float p0[8], p1[8], fac[8];
#pragma unroll
    for (int r = 0; r < 8; ++r) {
      float a0 = s0[r] * scale + mb0;
      float a1 = s1[r] * scale + mb1;
      float mx = fmaxf(a0, a1);
      for (int off = 1; off < 16; off <<= 1) mx = fmaxf(mx, __shfl_xor(mx, off, 16));
      float mn = fmaxf(m_run[r], mx);
      float f  = __expf(m_run[r] - mn);
      float e0 = __expf(a0 - mn);
      float e1 = __expf(a1 - mn);
      float rs = e0 + e1;
      for (int off = 1; off < 16; off <<= 1) rs += __shfl_xor(rs, off, 16);
      l_run[r] = l_run[r] * f + rs;
      m_run[r] = mn;
      fac[r] = f; p0[r] = e0; p1[r] = e1;
    }
#pragma unroll
    for (int c = 0; c < 4; ++c)
#pragma unroll
      for (int r = 0; r < 8; ++r) o[c][r] *= fac[r];

    // ---- restripe P: C-layout -> A-operand layout via per-wave LDS tile ----
#pragma unroll
    for (int r = 0; r < 8; ++r) {
      const int mrw = half * 8 + r;
      myP[mrw * 32 + ln]      = f2bf(p0[r]);
      myP[mrw * 32 + 16 + ln] = f2bf(p1[r]);
    }
    WAVE_LDS_SYNC();
    BF16x16 pa;
    pa.h[0] = *(const v8bf*)(myP + ln * 32 + half * 16);
    pa.h[1] = *(const v8bf*)(myP + ln * 32 + half * 16 + 8);

    // ---- O += P @ V (V^T layout -> contiguous-K B-operand loads) ----
#pragma unroll
    for (int c = 0; c < 4; ++c) {
      const bf16* vr = vbase + (size_t)(c * 16 + ln) * T_ + kb + half * 16;
      BF16x16 bv;
      bv.h[0] = *(const v8bf*)(vr);
      bv.h[1] = *(const v8bf*)(vr + 8);
      o[c] = wmma_bf16(pa, bv, o[c]);
    }
  };

  // double-buffered key loop: K-tile loads overlap softmax/PV of previous block
  KTile kt0, kt1;
  load_ktile(kt0, kbase, 0, ln, half);
  for (int kb = 0; kb < T_ - 64; kb += 64) {
    load_ktile(kt1, kbase, kb + 32, ln, half);
    body(kt0, kb);
    load_ktile(kt0, kbase, kb + 64, ln, half);
    body(kt1, kb + 32);
  }
  load_ktile(kt1, kbase, T_ - 32, ln, half);
  body(kt0, T_ - 64);
  body(kt1, T_ - 32);

  // ---- normalize and store ctx as bf16 [B,T,H] ----
#pragma unroll
  for (int c = 0; c < 4; ++c) {
#pragma unroll
    for (int r = 0; r < 8; ++r) {
      float val = o[c][r] / l_run[r];
      const size_t t   = (size_t)qt * 16 + half * 8 + r;
      const size_t col = (size_t)h * 64 + c * 16 + ln;
      ctx[((size_t)b * T_ + t) * H_ + col] = f2bf(val);
    }
  }
}

extern "C" void kernel_launch(void* const* d_in, const int* in_sizes, int n_in,
                              void* d_out, int out_size, void* d_ws, size_t ws_size,
                              hipStream_t stream) {
  (void)in_sizes; (void)n_in; (void)out_size; (void)ws_size;
  const float* x  = (const float*)d_in[0];
  const int*   am = (const int*)d_in[1];
  const float* Wq = (const float*)d_in[2];
  const float* bq = (const float*)d_in[3];
  const float* Wk = (const float*)d_in[4];
  const float* bk = (const float*)d_in[5];
  const float* Wv = (const float*)d_in[6];
  const float* bv = (const float*)d_in[7];
  const float* Wo = (const float*)d_in[8];
  const float* bo = (const float*)d_in[9];
  float* out = (float*)d_out;

  char* ws = (char*)d_ws;
  const size_t szX = (size_t)B_ * T_ * H_ * sizeof(bf16);   // 8 MB
  const size_t szW = (size_t)H_ * H_ * sizeof(bf16);        // 2 MB
  bf16* xb  = (bf16*)ws; ws += szX;
  bf16* WqT = (bf16*)ws; ws += szW;
  bf16* WkT = (bf16*)ws; ws += szW;
  bf16* WvT = (bf16*)ws; ws += szW;
  bf16* WoT = (bf16*)ws; ws += szW;
  bf16* qb  = (bf16*)ws; ws += szX;
  bf16* kb  = (bf16*)ws; ws += szX;
  bf16* vTb = (bf16*)ws; ws += szX;
  bf16* ctx = (bf16*)ws; ws += szX;

  const long nX = (long)B_ * T_ * H_;
  cvt_f32_bf16<<<2048, 256, 0, stream>>>(x, xb, nX);
  transpose_cvt<<<(H_ * H_) / 256, 256, 0, stream>>>(Wq, WqT);
  transpose_cvt<<<(H_ * H_) / 256, 256, 0, stream>>>(Wk, WkT);
  transpose_cvt<<<(H_ * H_) / 256, 256, 0, stream>>>(Wv, WvT);
  transpose_cvt<<<(H_ * H_) / 256, 256, 0, stream>>>(Wo, WoT);

  const int tiles = (B_ * T_ / 32) * (H_ / 64);   // 2048 wave tiles
  proj_gemm<<<tiles / 8, 256, 0, stream>>>(xb, WqT, bq, qb, 0);
  proj_gemm<<<tiles / 8, 256, 0, stream>>>(xb, WkT, bk, kb, 0);
  proj_gemm<<<tiles / 8, 256, 0, stream>>>(xb, WvT, bv, vTb, 1);

  flash_attn<<<(B_ * NH_ * (T_ / 16)) / 8, 256, 0, stream>>>(qb, kb, vTb, am, ctx);

  proj_gemm<<<tiles / 8, 256, 0, stream>>>(ctx, WoT, bo, out, 2);
}
